// TemporalGraphSAGE_14568529068621
// MI455X (gfx1250) — compile-verified, matrix operand order checked by reading, not run on hardware
//
#include <hip/hip_runtime.h>

// ---------------------------------------------------------------------------
// CDNA5 (gfx1250) WMMA types and fragment helpers
// ---------------------------------------------------------------------------
typedef __attribute__((ext_vector_type(16))) _Float16 v16h;
typedef __attribute__((ext_vector_type(8)))  _Float16 v8h;
typedef __attribute__((ext_vector_type(8)))  float    v8f;

__device__ __forceinline__ v8f wmma32(v16h a, v16h b, v8f c) {
  return __builtin_amdgcn_wmma_f32_16x16x32_f16(false, a, false, b, (short)0, c, false, false);
}

// A-fragment (16x32 f16) from a row-major source with leading dim lda (halfs).
// ISA 7.12.2: lane L holds row m = L&15; element j -> k = 16*(j>>3)+8*(L>>4)+(j&7).
// Per lane this is two contiguous 16-byte runs: [kh..kh+7] and [kh+16..kh+23].
__device__ __forceinline__ v16h frag_a(const _Float16* __restrict__ A, int lda, int lane) {
  const int m  = lane & 15;
  const int kh = (lane >> 4) << 3;
  const _Float16* p = A + m * lda + kh;
  v8h lo = *(const v8h*)(p);        // k = kh .. kh+7
  v8h hi = *(const v8h*)(p + 16);   // k = kh+16 .. kh+23
  return __builtin_shufflevector(lo, hi, 0, 1, 2, 3, 4, 5, 6, 7,
                                         8, 9, 10, 11, 12, 13, 14, 15);
}

// B-fragment from fragment-packed weights: layout [fb][lane][16] contiguous,
// so each lane does one 32-byte contiguous load (2x b128).
__device__ __forceinline__ v16h frag_packed(const _Float16* __restrict__ Bp, long fb, int lane) {
  const _Float16* p = Bp + (fb * 32 + lane) * 16;
  v8h lo = *(const v8h*)(p);
  v8h hi = *(const v8h*)(p + 8);
  return __builtin_shufflevector(lo, hi, 0, 1, 2, 3, 4, 5, 6, 7,
                                         8, 9, 10, 11, 12, 13, 14, 15);
}

// ---------------------------------------------------------------------------
// Weight repack: f32 [K,N] row slice -> fragment-packed f16.
// dst layout: [dstK/32][Ncols/16][32 lanes][16 elems], zero K-padded.
// ---------------------------------------------------------------------------
__global__ void pack_w_frag_kernel(const float* __restrict__ W, int rowOff, int srcK, int Ncols,
                                   _Float16* __restrict__ dst, int dstK) {
  int i = blockIdx.x * blockDim.x + threadIdx.x;
  int total = dstK * Ncols;
  if (i >= total) return;
  int j    = i & 15;
  int lane = (i >> 4) & 31;
  int fb   = i >> 9;                 // kb*ntiles + nt
  int ntiles = Ncols >> 4;
  int kb = fb / ntiles;
  int nt = fb - kb * ntiles;
  int n = nt * 16 + (lane & 15);
  int k = kb * 32 + ((j >> 3) << 4) + ((lane >> 4) << 3) + (j & 7);
  float v = (k < srcK) ? W[(long)(rowOff + k) * Ncols + n] : 0.f;
  dst[i] = (_Float16)v;
}

// ---------------------------------------------------------------------------
// LayerNorm over 130 features, write f16 zero-padded to 160 cols
// ---------------------------------------------------------------------------
__global__ void ln_kernel(const float* __restrict__ x, const float* __restrict__ g,
                          const float* __restrict__ b, _Float16* __restrict__ xln, int Nn) {
  int lane = threadIdx.x & 31;
  int node = (blockIdx.x * blockDim.x + threadIdx.x) >> 5;
  if (node >= Nn) return;
  const float* row = x + (long)node * 130;
  float s = 0.f, s2 = 0.f;
  for (int i = lane; i < 130; i += 32) { float v = row[i]; s += v; s2 += v * v; }
#pragma unroll
  for (int o = 16; o > 0; o >>= 1) { s += __shfl_down(s, o, 32); s2 += __shfl_down(s2, o, 32); }
  s = __shfl(s, 0, 32); s2 = __shfl(s2, 0, 32);
  float mu  = s / 130.f;
  float var = s2 / 130.f - mu * mu;
  float inv = rsqrtf(var + 1e-5f);
  _Float16* orow = xln + (long)node * 160;
  for (int i = lane; i < 160; i += 32) {
    float v = (i < 130) ? (row[i] - mu) * inv * g[i] + b[i] : 0.f;
    orow[i] = (_Float16)v;
  }
}

// ---------------------------------------------------------------------------
// Generic WMMA GEMM: C[M,N] = A[M,K] @ Bpacked + bias, optional relu.
// 8 waves / block, one 16x16 tile per wave. M must be a multiple of 16.
// ---------------------------------------------------------------------------
template <bool RELU, bool WF, bool WH>
__global__ __launch_bounds__(256) void
gemm_f16_kernel(const _Float16* __restrict__ A, int lda,
                const _Float16* __restrict__ Bp, const float* __restrict__ bias,
                int M, int K, int NC,
                float* __restrict__ Cf, _Float16* __restrict__ Cb, int ldc) {
  const int lane = threadIdx.x & 31;
  const int wave = threadIdx.x >> 5;
  const int ntiles = NC >> 4;
  const long tile = (long)blockIdx.x * 8 + wave;
  const long mt = tile / ntiles;
  const int  nt = (int)(tile - mt * ntiles);
  if (mt * 16 >= M) return;                 // wave-uniform
  const _Float16* Ab = A + mt * 16 * (long)lda;
  v8f acc = {};
  const int kblocks = K >> 5;
  for (int kb = 0; kb < kblocks; ++kb) {
    v16h af = frag_a(Ab + kb * 32, lda, lane);
    v16h bf = frag_packed(Bp, (long)kb * ntiles + nt, lane);
    acc = wmma32(af, bf, acc);
  }
  const int n  = nt * 16 + (lane & 15);
  const float bv = bias ? bias[n] : 0.f;
#pragma unroll
  for (int r = 0; r < 8; ++r) {
    const int m = (int)(mt * 16) + r + ((lane >> 4) << 3);
    float v = acc[r] + bv;
    if (RELU) v = fmaxf(v, 0.f);
    if (WF) Cf[(long)m * ldc + n] = v;
    if (WH) Cb[(long)m * ldc + n] = (_Float16)v;
  }
}

template <bool RELU, bool WF, bool WH>
static inline void launch_gemm(hipStream_t stream, const _Float16* A, int lda,
                               const _Float16* Bp, const float* bias,
                               int M, int K, int NC, float* Cf, _Float16* Cb, int ldc) {
  long tiles = (long)((M + 15) / 16) * (NC / 16);
  int blocks = (int)((tiles + 7) / 8);
  gemm_f16_kernel<RELU, WF, WH><<<blocks, 256, 0, stream>>>(A, lda, Bp, bias, M, K, NC, Cf, Cb, ldc);
}

// ---------------------------------------------------------------------------
// Edge message kernel (hot loop). 16 edges / block, 8 waves.
// Stage A: pre = relu(eattr@We + P[dst] + Q[src] + b1)   (one WMMA K-step)
// Stage B: msg = pre @ W2 + b2 ; atomicAdd scatter to S[dst]
// We and W2 are fragment-packed.
// ---------------------------------------------------------------------------
template <int NC>
__global__ __launch_bounds__(256) void
edge_msg_kernel(const int* __restrict__ src, const int* __restrict__ dst,
                const float* __restrict__ eattr,
                const _Float16* __restrict__ We,   // packed [1][NC/16][32][16]
                const float* __restrict__ b1,
                const float* __restrict__ P,       // [N][NC]
                const float* __restrict__ Q,       // [N][NC]
                const _Float16* __restrict__ W2,   // packed [NC/32][NC/16][32][16]
                const float* __restrict__ b2,
                float* __restrict__ S, int E) {
  __shared__ int sdst[16], ssrc[16];
  __shared__ __align__(16) _Float16 se[16][32];
  __shared__ __align__(16) _Float16 pre[16][NC];
  const int tid  = threadIdx.x;
  const int lane = tid & 31;
  const int wave = tid >> 5;
  const int ntiles = NC / 16;
  const long e0 = (long)blockIdx.x * 16;

  if (tid < 16) {
    long ee = e0 + tid; if (ee >= E) ee = E - 1;
    sdst[tid] = dst[ee];
  } else if (tid < 32) {
    long ee = e0 + (tid - 16); if (ee >= E) ee = E - 1;
    ssrc[tid - 16] = src[ee];
  }
  for (int s = tid; s < 16 * 32; s += 256) {
    int r = s >> 5, c = s & 31;
    long ee = e0 + r;
    float v = (c < 27 && ee < E) ? eattr[ee * 27 + c] : 0.f;
    se[r][c] = (_Float16)v;
  }
  __syncthreads();

  // Stage A
  for (int t = wave; t < ntiles; t += 8) {
    v8f acc = {};
    v16h af = frag_a(&se[0][0], 32, lane);
    v16h bf = frag_packed(We, t, lane);
    acc = wmma32(af, bf, acc);
    const int n = t * 16 + (lane & 15);
#pragma unroll
    for (int r = 0; r < 8; ++r) {
      const int m = r + ((lane >> 4) << 3);
      const long d = sdst[m], sj = ssrc[m];
      float v = acc[r] + P[d * NC + n] + Q[sj * NC + n] + b1[n];
      pre[m][n] = (_Float16)fmaxf(v, 0.f);
    }
  }
  __syncthreads();

  // Stage B
  for (int t = wave; t < ntiles; t += 8) {
    v8f acc = {};
#pragma unroll
    for (int k = 0; k < NC; k += 32) {
      v16h af = frag_a(&pre[0][k], NC, lane);
      v16h bf = frag_packed(W2, (long)(k >> 5) * ntiles + t, lane);
      acc = wmma32(af, bf, acc);
    }
    const int n = t * 16 + (lane & 15);
#pragma unroll
    for (int r = 0; r < 8; ++r) {
      const int m = r + ((lane >> 4) << 3);
      if (e0 + m >= E) continue;
      const long d = sdst[m];
      atomicAdd(&S[d * NC + n], acc[r] + b2[n]);
    }
  }
}

// ---------------------------------------------------------------------------
// Elementwise helpers
// ---------------------------------------------------------------------------
__global__ void zero_f32(float* p, long n) {
  long i = (long)blockIdx.x * blockDim.x + threadIdx.x;
  if (i < n) p[i] = 0.f;
}

__global__ void deg_kernel(const int* __restrict__ dst, float* __restrict__ deg, int E) {
  int e = blockIdx.x * blockDim.x + threadIdx.x;
  if (e < E) atomicAdd(&deg[dst[e]], 1.0f);
}

// cat[n] = [ h_f16[n] (HC) | f16(S[n]/max(deg,1)) (AC) ]
__global__ void concat_agg_kernel(const _Float16* __restrict__ hb, const float* __restrict__ S,
                                  const float* __restrict__ deg, _Float16* __restrict__ cat,
                                  int Nn, int HC, int AC) {
  long i = (long)blockIdx.x * blockDim.x + threadIdx.x;
  long total = (long)Nn * (HC + AC);
  if (i >= total) return;
  int n = (int)(i / (HC + AC)), c = (int)(i - (long)n * (HC + AC));
  _Float16 v;
  if (c < HC) v = hb[(long)n * HC + c];
  else {
    float d = fmaxf(deg[n], 1.f);
    v = (_Float16)(S[(long)n * AC + (c - HC)] / d);
  }
  cat[i] = v;
}

// h += relu(bn(h1)); mirror f16
__global__ void bn_relu_add_kernel(const float* __restrict__ h1,
                                   const float* __restrict__ g, const float* __restrict__ b,
                                   const float* __restrict__ m, const float* __restrict__ v,
                                   float* __restrict__ hf, _Float16* __restrict__ hb,
                                   int Nn, int C) {
  long i = (long)blockIdx.x * blockDim.x + threadIdx.x;
  if (i >= (long)Nn * C) return;
  int c = (int)(i % C);
  float t = (h1[i] - m[c]) * rsqrtf(v[c] + 1e-5f) * g[c] + b[c];
  t = fmaxf(t, 0.f);
  float o = hf[i] + t;
  hf[i] = o;
  hb[i] = (_Float16)o;
}

// h3 = R + relu(bn(h2)); mirror f16
__global__ void bn_relu_res_kernel(const float* __restrict__ h2, const float* __restrict__ R,
                                   const float* __restrict__ g, const float* __restrict__ b,
                                   const float* __restrict__ m, const float* __restrict__ v,
                                   float* __restrict__ h3f, _Float16* __restrict__ h3b,
                                   int Nn, int C) {
  long i = (long)blockIdx.x * blockDim.x + threadIdx.x;
  if (i >= (long)Nn * C) return;
  int c = (int)(i % C);
  float t = (h2[i] - m[c]) * rsqrtf(v[c] + 1e-5f) * g[c] + b[c];
  t = fmaxf(t, 0.f);
  float o = R[i] + t;
  h3f[i] = o;
  h3b[i] = (_Float16)o;
}

// logits[n] = t[n,0:32] . W2 + b2
__global__ void cls_kernel(const float* __restrict__ t, const float* __restrict__ W2,
                           const float* __restrict__ b2, float* __restrict__ out, int Nn) {
  int n = blockIdx.x * blockDim.x + threadIdx.x;
  if (n >= Nn) return;
  float s = b2[0];
  const float* row = t + (long)n * 32;
#pragma unroll
  for (int i = 0; i < 32; ++i) s += row[i] * W2[i];
  out[n] = s;
}

// ---------------------------------------------------------------------------
// Host launcher
// ---------------------------------------------------------------------------
extern "C" void kernel_launch(void* const* d_in, const int* in_sizes, int n_in,
                              void* d_out, int out_size, void* d_ws, size_t ws_size,
                              hipStream_t stream) {
  const int ND = 130, ED = 27;
  const int Nn = in_sizes[0] / ND;
  const int E  = in_sizes[1] / ED;

  const float* x       = (const float*)d_in[0];
  const float* eattr   = (const float*)d_in[1];
  const float* ln_g    = (const float*)d_in[2];
  const float* ln_b    = (const float*)d_in[3];
  const float* proj_W  = (const float*)d_in[4];
  const float* proj_b  = (const float*)d_in[5];
  const float* c1_mW1  = (const float*)d_in[6];
  const float* c1_mb1  = (const float*)d_in[7];
  const float* c1_mW2  = (const float*)d_in[8];
  const float* c1_mb2  = (const float*)d_in[9];
  const float* c1_uW1  = (const float*)d_in[10];
  const float* c1_ub1  = (const float*)d_in[11];
  const float* c1_uW2  = (const float*)d_in[12];
  const float* c1_ub2  = (const float*)d_in[13];
  const float* bn1_g   = (const float*)d_in[14];
  const float* bn1_b   = (const float*)d_in[15];
  const float* bn1_m   = (const float*)d_in[16];
  const float* bn1_v   = (const float*)d_in[17];
  const float* c2_mW1  = (const float*)d_in[18];
  const float* c2_mb1  = (const float*)d_in[19];
  const float* c2_mW2  = (const float*)d_in[20];
  const float* c2_mb2  = (const float*)d_in[21];
  const float* c2_uW1  = (const float*)d_in[22];
  const float* c2_ub1  = (const float*)d_in[23];
  const float* c2_uW2  = (const float*)d_in[24];
  const float* c2_ub2  = (const float*)d_in[25];
  const float* bn2_g   = (const float*)d_in[26];
  const float* bn2_b   = (const float*)d_in[27];
  const float* bn2_m   = (const float*)d_in[28];
  const float* bn2_v   = (const float*)d_in[29];
  const float* res2_W  = (const float*)d_in[30];
  const float* res2_b  = (const float*)d_in[31];
  const float* cls_W1  = (const float*)d_in[32];
  const float* cls_b1  = (const float*)d_in[33];
  const float* cls_W2  = (const float*)d_in[34];
  const float* cls_b2  = (const float*)d_in[35];
  const int*   ei      = (const int*)d_in[36];
  const int*   srcI    = ei;
  const int*   dstI    = ei + E;
  float*       out     = (float*)d_out;

  char* ws = (char*)d_ws;
  size_t off = 0;
  auto alloc = [&](size_t bytes) -> void* {
    void* p = ws + off;
    off += (bytes + 255) & ~(size_t)255;
    return p;
  };

  // fragment-packed f16 weights (zero K-padded to multiples of 32)
  _Float16* wproj = (_Float16*)alloc((size_t)160 * 128 * 2);
  _Float16* w1d   = (_Float16*)alloc((size_t)128 * 128 * 2);
  _Float16* w1s   = (_Float16*)alloc((size_t)128 * 128 * 2);
  _Float16* w1e   = (_Float16*)alloc((size_t)32  * 128 * 2);
  _Float16* w1m2  = (_Float16*)alloc((size_t)128 * 128 * 2);
  _Float16* w1u1  = (_Float16*)alloc((size_t)256 * 128 * 2);
  _Float16* w1u2  = (_Float16*)alloc((size_t)128 * 128 * 2);
  _Float16* w2d   = (_Float16*)alloc((size_t)128 * 64 * 2);
  _Float16* w2s   = (_Float16*)alloc((size_t)128 * 64 * 2);
  _Float16* w2e   = (_Float16*)alloc((size_t)32  * 64 * 2);
  _Float16* w2m2  = (_Float16*)alloc((size_t)64  * 64 * 2);
  _Float16* w2u1  = (_Float16*)alloc((size_t)192 * 64 * 2);
  _Float16* w2u2  = (_Float16*)alloc((size_t)64  * 64 * 2);
  _Float16* wres  = (_Float16*)alloc((size_t)128 * 64 * 2);
  _Float16* wcls  = (_Float16*)alloc((size_t)64  * 32 * 2);

  // f16 activations
  _Float16* xln  = (_Float16*)alloc((size_t)Nn * 160 * 2);
  _Float16* hb   = (_Float16*)alloc((size_t)Nn * 128 * 2);
  _Float16* cat  = (_Float16*)alloc((size_t)Nn * 256 * 2);   // reused [N,192] for conv2
  _Float16* uact = (_Float16*)alloc((size_t)Nn * 128 * 2);   // reused [N,64]
  _Float16* h3b  = (_Float16*)alloc((size_t)Nn * 64 * 2);

  // f32 buffers
  float* hf  = (float*)alloc((size_t)Nn * 128 * 4);
  float* P   = (float*)alloc((size_t)Nn * 128 * 4);
  float* Q   = (float*)alloc((size_t)Nn * 128 * 4);
  float* S   = (float*)alloc((size_t)Nn * 128 * 4);
  float* deg = (float*)alloc((size_t)Nn * 4);
  // aliases with disjoint lifetimes
  float* h1buf = S;
  float* h2    = S;
  float* R     = S + (size_t)Nn * 64;
  float* h3f   = P;
  float* tcls  = Q;

  auto packw = [&](const float* Wsrc, int rowOff, int srcK, int Ncols, _Float16* dp, int dstK) {
    int total = dstK * Ncols;
    pack_w_frag_kernel<<<(total + 255) / 256, 256, 0, stream>>>(Wsrc, rowOff, srcK, Ncols, dp, dstK);
  };
  auto zero = [&](float* p, long n) {
    zero_f32<<<(int)((n + 255) / 256), 256, 0, stream>>>(p, n);
  };

  // -- weight repack ---------------------------------------------------------
  packw(proj_W, 0, 130, 128, wproj, 160);
  packw(c1_mW1, 0,   128, 128, w1d, 128);
  packw(c1_mW1, 128, 128, 128, w1s, 128);
  packw(c1_mW1, 256, 27,  128, w1e, 32);
  packw(c1_mW2, 0, 128, 128, w1m2, 128);
  packw(c1_uW1, 0, 256, 128, w1u1, 256);
  packw(c1_uW2, 0, 128, 128, w1u2, 128);
  packw(c2_mW1, 0,   128, 64, w2d, 128);
  packw(c2_mW1, 128, 128, 64, w2s, 128);
  packw(c2_mW1, 256, 27,  64, w2e, 32);
  packw(c2_mW2, 0, 64, 64, w2m2, 64);
  packw(c2_uW1, 0, 192, 64, w2u1, 192);
  packw(c2_uW2, 0, 64, 64, w2u2, 64);
  packw(res2_W, 0, 128, 64, wres, 128);
  packw(cls_W1, 0, 64, 32, wcls, 64);

  // -- LN + projection -------------------------------------------------------
  ln_kernel<<<(Nn * 32 + 255) / 256, 256, 0, stream>>>(x, ln_g, ln_b, xln, Nn);
  launch_gemm<true, true, true>(stream, xln, 160, wproj, proj_b, Nn, 160, 128, hf, hb, 128);

  // -- conv1 -----------------------------------------------------------------
  launch_gemm<false, true, false>(stream, hb, 128, w1d, nullptr, Nn, 128, 128, P, nullptr, 128);
  launch_gemm<false, true, false>(stream, hb, 128, w1s, nullptr, Nn, 128, 128, Q, nullptr, 128);
  zero(S, (long)Nn * 128);
  zero(deg, Nn);
  deg_kernel<<<(E + 255) / 256, 256, 0, stream>>>(dstI, deg, E);
  edge_msg_kernel<128><<<(E + 15) / 16, 256, 0, stream>>>(srcI, dstI, eattr, w1e, c1_mb1,
                                                          P, Q, w1m2, c1_mb2, S, E);
  {
    long tot = (long)Nn * 256;
    concat_agg_kernel<<<(int)((tot + 255) / 256), 256, 0, stream>>>(hb, S, deg, cat, Nn, 128, 128);
  }
  launch_gemm<true, false, true>(stream, cat, 256, w1u1, c1_ub1, Nn, 256, 128, nullptr, uact, 128);
  launch_gemm<false, true, false>(stream, uact, 128, w1u2, c1_ub2, Nn, 128, 128, h1buf, nullptr, 128);
  {
    long tot = (long)Nn * 128;
    bn_relu_add_kernel<<<(int)((tot + 255) / 256), 256, 0, stream>>>(h1buf, bn1_g, bn1_b, bn1_m,
                                                                     bn1_v, hf, hb, Nn, 128);
  }

  // -- conv2 -----------------------------------------------------------------
  launch_gemm<false, true, false>(stream, hb, 128, w2d, nullptr, Nn, 128, 64, P, nullptr, 64);
  launch_gemm<false, true, false>(stream, hb, 128, w2s, nullptr, Nn, 128, 64, Q, nullptr, 64);
  zero(S, (long)Nn * 64);
  zero(deg, Nn);
  deg_kernel<<<(E + 255) / 256, 256, 0, stream>>>(dstI, deg, E);
  edge_msg_kernel<64><<<(E + 15) / 16, 256, 0, stream>>>(srcI, dstI, eattr, w2e, c2_mb1,
                                                         P, Q, w2m2, c2_mb2, S, E);
  {
    long tot = (long)Nn * 192;
    concat_agg_kernel<<<(int)((tot + 255) / 256), 256, 0, stream>>>(hb, S, deg, cat, Nn, 128, 64);
  }
  launch_gemm<true, false, true>(stream, cat, 192, w2u1, c2_ub1, Nn, 192, 64, nullptr, uact, 64);
  launch_gemm<false, true, false>(stream, uact, 64, w2u2, c2_ub2, Nn, 64, 64, h2, nullptr, 64);
  launch_gemm<false, true, false>(stream, hb, 128, wres, res2_b, Nn, 128, 64, R, nullptr, 64);
  {
    long tot = (long)Nn * 64;
    bn_relu_res_kernel<<<(int)((tot + 255) / 256), 256, 0, stream>>>(h2, R, bn2_g, bn2_b, bn2_m,
                                                                     bn2_v, h3f, h3b, Nn, 64);
  }

  // -- classifier ------------------------------------------------------------
  launch_gemm<true, true, false>(stream, h3b, 64, wcls, cls_b1, Nn, 64, 32, tcls, nullptr, 32);
  cls_kernel<<<(Nn + 255) / 256, 256, 0, stream>>>(tcls, cls_W2, cls_b2, out, Nn);
}
